// PaiNNBlock_17514876634211
// MI455X (gfx1250) — matrix operand order repaired and checked by reading, not compiled, not run on hardware
//
#include <hip/hip_runtime.h>
#include <math.h>

typedef __attribute__((ext_vector_type(16))) _Float16 v16h;
typedef __attribute__((ext_vector_type(8)))  float    v8f;

#define H   128
#define H3  384
#define RBF 20
#define ET  32          // edges per block in the fused edge kernel

__device__ __forceinline__ float silu_f(float v) {
    return v * (1.0f / (1.0f + __expf(-v)));
}

// ---------------------------------------------------------------------------
// Pack W [K,N] f32 row-major into WMMA B-fragment order (f16).
// Fragment (nt,kt): lane n<16 holds col nt*16+n, K = kt*32 + 0..7, 16..23
//                   lane n>=16 holds col nt*16+(n-16), K = kt*32 + 8..15, 24..31
// ---------------------------------------------------------------------------
__global__ void pack_b_kernel(const float* __restrict__ W, _Float16* __restrict__ Wp,
                              int K, int N) {
    const int Kt   = K >> 5;
    const int frag = blockIdx.x;          // frag = nt*Kt + kt
    const int kt   = frag % Kt;
    const int nt   = frag / Kt;
    const int lane = threadIdx.x & 31;
    const int n     = nt * 16 + (lane & 15);
    const int kbase = kt * 32 + ((lane < 16) ? 0 : 8);
    _Float16* dst = Wp + ((size_t)frag * 32 + lane) * 16;
#pragma unroll
    for (int j = 0; j < 16; ++j)
        dst[j] = (_Float16)W[(size_t)(kbase + (j < 8 ? j : j + 8)) * N + n];
}

// ---------------------------------------------------------------------------
// Generic WMMA GEMM: C[M,N] = epilogue(A[M,K] @ W[K,N] + bias)
// Software-pipelined: A f32 quads and B fragment loaded one kt ahead.
// block = 128 (4 waves); wave owns one 16x16 C tile.
// grid.x = M/16, grid.y = ceil(N/64). M%16==0, N%16==0, K%32==0.
// ---------------------------------------------------------------------------
template <int EPI>   // 0 = identity, 1 = SiLU
__global__ __launch_bounds__(128)
void wmma_gemm_kernel(const float* __restrict__ A, const _Float16* __restrict__ Bp,
                      const float* __restrict__ bias, float* __restrict__ C,
                      int M, int N, int K) {
    const int wave  = threadIdx.x >> 5;
    const int lane  = threadIdx.x & 31;
    const int col   = lane & 15;
    const int rbase = (lane < 16) ? 0 : 8;
    const int ks0   = (lane < 16) ? 0 : 8;
    const int m0    = blockIdx.x * 16;
    const int nt    = blockIdx.y * 4 + wave;
    const int n0    = nt * 16;
    if (m0 >= M || n0 >= N) return;          // wave-uniform
    const int Kt = K >> 5;

    v8f acc;
    const float b = bias ? bias[n0 + col] : 0.0f;
#pragma unroll
    for (int i = 0; i < 8; ++i) acc[i] = b;

    const float*    arow  = A + (size_t)(m0 + col) * K;
    const _Float16* bfrag = Bp + ((size_t)nt * Kt * 32 + lane) * 16;

    // prologue loads (kt = 0)
    float4 a0 = *(const float4*)(arow + ks0 + 0);
    float4 a1 = *(const float4*)(arow + ks0 + 4);
    float4 a2 = *(const float4*)(arow + ks0 + 16);
    float4 a3 = *(const float4*)(arow + ks0 + 20);
    v16h  bcur = *(const v16h*)bfrag;

    for (int kt = 0; kt < Kt; ++kt) {
        float4 na0 = a0, na1 = a1, na2 = a2, na3 = a3;
        v16h   bnext = bcur;
        if (kt + 1 < Kt) {                   // prefetch next iteration
            const float* ap = arow + (kt + 1) * 32 + ks0;
            na0 = *(const float4*)(ap + 0);
            na1 = *(const float4*)(ap + 4);
            na2 = *(const float4*)(ap + 16);
            na3 = *(const float4*)(ap + 20);
            bnext = *(const v16h*)(bfrag + 512);
        }
        v16h a;
        a[0]  = (_Float16)a0.x; a[1]  = (_Float16)a0.y; a[2]  = (_Float16)a0.z; a[3]  = (_Float16)a0.w;
        a[4]  = (_Float16)a1.x; a[5]  = (_Float16)a1.y; a[6]  = (_Float16)a1.z; a[7]  = (_Float16)a1.w;
        a[8]  = (_Float16)a2.x; a[9]  = (_Float16)a2.y; a[10] = (_Float16)a2.z; a[11] = (_Float16)a2.w;
        a[12] = (_Float16)a3.x; a[13] = (_Float16)a3.y; a[14] = (_Float16)a3.z; a[15] = (_Float16)a3.w;

        acc = __builtin_amdgcn_wmma_f32_16x16x32_f16(false, a, false, bcur,
                                                     (short)0, acc, false, false);
        a0 = na0; a1 = na1; a2 = na2; a3 = na3;
        bcur = bnext;
        bfrag += 512;
    }

#pragma unroll
    for (int i = 0; i < 8; ++i) {
        float v = acc[i];
        if (EPI == 1) v = silu_f(v);
        C[(size_t)(m0 + rbase + i) * N + (n0 + col)] = v;
    }
}

// ---------------------------------------------------------------------------
// Fused edge kernel. One 128-thread block (4 waves) handles ET=32 edges
// (2 WMMA M-tiles sharing each B fragment):
//  stage0: hf = silu(rbf @ Wf1 + bf1)          (VALU, K=20) -> LDS
//  stage1: filters = (hf @ Wf2 + bf2)*cutoff   (WMMA; A frags register-resident,
//          4 B frags preloaded per n-tile), scalar msgs via atomics,
//          r/mu filters * gathered x -> LDS
//  stage2: vector messages: atomicAdd mumsg (unit*xr + mu_src*xm)
// Requires E % ET == 0 (E = 320000).
// ---------------------------------------------------------------------------
__global__ __launch_bounds__(128)
void edge_kernel(const float* __restrict__ x,       // [N,384]
                 const float* __restrict__ mu,      // [N,3,128]
                 const int*   __restrict__ tgt,     // edge_index[0]
                 const int*   __restrict__ src,     // edge_index[1]
                 const float* __restrict__ rbf,     // [E,20]
                 const float* __restrict__ unitv,   // [E,3]
                 const float* __restrict__ cut,     // [E]
                 const float* __restrict__ Wf1,     // [20,128]
                 const float* __restrict__ bf1,     // [128]
                 const _Float16* __restrict__ Wf2p, // packed (Kt=4, Nt=24)
                 const float* __restrict__ bf2,     // [384]
                 float* __restrict__ qmsg,          // [N,128]
                 float* __restrict__ mumsg,         // [N,3,128]
                 float* __restrict__ deg)           // [N]
{
    __shared__ float sWf1[RBF * H];
    __shared__ float sbf1[H];
    __shared__ float sRbf[ET * RBF];
    __shared__ float sA [ET * H];      // hf tile (post-SiLU)
    __shared__ float sFr[ET * H];      // x_r[src]*filter_r
    __shared__ float sFm[ET * H];      // x_mu[src]*filter_mu
    __shared__ float sCut[ET];
    __shared__ int   sSrc[ET], sTgt[ET];
    __shared__ float sUnit[ET * 3];

    const int tid = threadIdx.x;
    const int e0  = blockIdx.x * ET;

    for (int i = tid; i < RBF * H; i += 128) sWf1[i] = Wf1[i];
    if (tid < H) sbf1[tid] = bf1[tid];
    for (int i = tid; i < ET * RBF; i += 128) sRbf[i] = rbf[(size_t)e0 * RBF + i];
    if (tid < ET) {
        sSrc[tid] = src[e0 + tid];
        sTgt[tid] = tgt[e0 + tid];
        sCut[tid] = cut[e0 + tid];
    }
    if (tid < ET * 3) sUnit[tid] = unitv[(size_t)e0 * 3 + tid];
    __syncthreads();

    // ---- stage0: filter MLP layer 1 (K=20, VALU) ----
    {
        const int h = tid;             // 0..127
        for (int m = 0; m < ET; ++m) {
            float acc = sbf1[h];
#pragma unroll
            for (int r = 0; r < RBF; ++r) acc += sRbf[m * RBF + r] * sWf1[r * H + h];
            sA[m * H + h] = silu_f(acc);
        }
    }
    __syncthreads();

    const int wave  = tid >> 5;
    const int lane  = tid & 31;
    const int col   = lane & 15;
    const int rbase = (lane < 16) ? 0 : 8;
    const int ks0   = (lane < 16) ? 0 : 8;

    // A fragments for both M-tiles, register-resident for the whole n-loop
    v16h aF[2][4];
#pragma unroll
    for (int mt = 0; mt < 2; ++mt)
#pragma unroll
        for (int kt = 0; kt < 4; ++kt) {
            const float* ap = &sA[(mt * 16 + col) * H + kt * 32 + ks0];
            v16h a;
#pragma unroll
            for (int j = 0; j < 8; ++j) {
                a[j]     = (_Float16)ap[j];
                a[8 + j] = (_Float16)ap[16 + j];
            }
            aF[mt][kt] = a;
        }

    for (int nt = wave; nt < 24; nt += 4) {
        const int n0 = nt * 16;

        // preload all 4 B fragments (4KB contiguous) -> clauseable loads
        v16h bF[4];
#pragma unroll
        for (int kt = 0; kt < 4; ++kt)
            bF[kt] = *(const v16h*)(Wf2p + ((size_t)(nt * 4 + kt) * 32 + lane) * 16);

        const float b = bf2[n0 + col];
        v8f acc0, acc1;
#pragma unroll
        for (int i = 0; i < 8; ++i) { acc0[i] = b; acc1[i] = b; }

#pragma unroll
        for (int kt = 0; kt < 4; ++kt)
            acc0 = __builtin_amdgcn_wmma_f32_16x16x32_f16(false, aF[0][kt], false, bF[kt],
                                                          (short)0, acc0, false, false);
#pragma unroll
        for (int kt = 0; kt < 4; ++kt)
            acc1 = __builtin_amdgcn_wmma_f32_16x16x32_f16(false, aF[1][kt], false, bF[kt],
                                                          (short)0, acc1, false, false);

#pragma unroll
        for (int mt = 0; mt < 2; ++mt) {
            const v8f acc = mt ? acc1 : acc0;
            const int mb  = mt * 16 + rbase;
            if (n0 < H) {                           // filter_q -> scalar messages
#pragma unroll
                for (int i = 0; i < 8; ++i) {
                    const int m = mb + i;
                    const float f = acc[i] * sCut[m];
                    const int s = sSrc[m];
                    const float v = x[(size_t)s * H3 + n0 + col] * f;
                    atomicAdd(&qmsg[(size_t)sTgt[m] * H + n0 + col], v);
                }
            } else if (n0 < 2 * H) {                // filter_r * x_r[src] -> LDS
#pragma unroll
                for (int i = 0; i < 8; ++i) {
                    const int m = mb + i;
                    const float f = acc[i] * sCut[m];
                    const int s = sSrc[m];
                    sFr[m * H + (n0 - H) + col] = x[(size_t)s * H3 + n0 + col] * f;
                }
            } else {                                // filter_mu * x_mu[src] -> LDS
#pragma unroll
                for (int i = 0; i < 8; ++i) {
                    const int m = mb + i;
                    const float f = acc[i] * sCut[m];
                    const int s = sSrc[m];
                    sFm[m * H + (n0 - 2 * H) + col] = x[(size_t)s * H3 + n0 + col] * f;
                }
            }
        }
    }
    __syncthreads();

    // ---- stage2: vector messages ----
    for (int idx = tid; idx < ET * H; idx += 128) {
        const int m = idx >> 7, h = idx & (H - 1);
        const int s = sSrc[m], t = sTgt[m];
        const float xr = sFr[idx], xm = sFm[idx];
#pragma unroll
        for (int d = 0; d < 3; ++d) {
            const float v = sUnit[m * 3 + d] * xr + mu[(size_t)s * H3 + d * H + h] * xm;
            atomicAdd(&mumsg[(size_t)t * H3 + d * H + h], v);
        }
    }
    if (tid < ET) atomicAdd(&deg[sTgt[tid]], 1.0f);
}

// ---------------------------------------------------------------------------
__global__ void zero_kernel(float* __restrict__ p, size_t n) {
    size_t i = (size_t)blockIdx.x * blockDim.x + threadIdx.x;
    if (i < n) p[i] = 0.0f;
}

__global__ void finalize_nodes_kernel(const float* __restrict__ q, const float* __restrict__ mu,
                                      const float* __restrict__ qmsg, const float* __restrict__ mumsg,
                                      const float* __restrict__ deg,
                                      float* __restrict__ qn, float* __restrict__ mun, int N) {
    size_t i  = (size_t)blockIdx.x * blockDim.x + threadIdx.x;
    size_t nq = (size_t)N * H;
    size_t nm = (size_t)N * H3;
    if (i < nq) {
        float d = fmaxf(deg[i / H], 1.0f);
        qn[i] = q[i] + qmsg[i] / d;
    } else if (i < nq + nm) {
        size_t j = i - nq;
        float d = fmaxf(deg[j / H3], 1.0f);
        mun[j] = mu[j] + mumsg[j] / d;
    }
}

__global__ void mix_reduce_kernel(const float* __restrict__ qn, const float* __restrict__ mu_cat,
                                  float* __restrict__ sinb, float* __restrict__ inner, int N) {
    int i = blockIdx.x * blockDim.x + threadIdx.x;   // over N*H
    if (i >= N * H) return;
    const int n = i / H, h = i % H;
    const float* mc = mu_cat + (size_t)n * 3 * 256;
    float s2 = 0.0f, in = 0.0f;
#pragma unroll
    for (int d = 0; d < 3; ++d) {
        float v = mc[d * 256 + h];
        float w = mc[d * 256 + 128 + h];
        s2 += v * v;
        in += v * w;
    }
    sinb[(size_t)n * 256 + h]       = qn[i];
    sinb[(size_t)n * 256 + 128 + h] = sqrtf(s2 + 1e-8f);
    inner[i] = in;
}

__global__ void final_update_kernel(float* __restrict__ qn, float* __restrict__ mun,
                                    const float* __restrict__ delta, const float* __restrict__ mu_cat,
                                    const float* __restrict__ inner, int N) {
    int i = blockIdx.x * blockDim.x + threadIdx.x;   // over N*H
    if (i >= N * H) return;
    const int n = i / H, h = i % H;
    const float* dl = delta + (size_t)n * H3;
    qn[i] += dl[h] + dl[2 * H + h] * inner[i];       // dq + dqmu*inner
    const float dscale = dl[H + h];                  // dmu_scale
    const float* mc = mu_cat + (size_t)n * 3 * 256;
    float* mp = mun + (size_t)n * H3;
#pragma unroll
    for (int d = 0; d < 3; ++d)
        mp[d * H + h] += mc[d * 256 + 128 + h] * dscale;   // mu_w * dmu_scale
}

// ---------------------------------------------------------------------------
extern "C" void kernel_launch(void* const* d_in, const int* in_sizes, int n_in,
                              void* d_out, int out_size, void* d_ws, size_t ws_size,
                              hipStream_t stream) {
    const float* q    = (const float*)d_in[0];
    const float* mu   = (const float*)d_in[1];
    const int*   ei   = (const int*)  d_in[2];
    const float* rbf  = (const float*)d_in[3];
    const float* unit = (const float*)d_in[4];
    const float* cut  = (const float*)d_in[5];
    const float* Wi1  = (const float*)d_in[6];
    const float* bi1  = (const float*)d_in[7];
    const float* Wi2  = (const float*)d_in[8];
    const float* bi2  = (const float*)d_in[9];
    const float* Wf1  = (const float*)d_in[10];
    const float* bf1  = (const float*)d_in[11];
    const float* Wf2  = (const float*)d_in[12];
    const float* bf2  = (const float*)d_in[13];
    const float* Wv   = (const float*)d_in[14];
    const float* Ws1  = (const float*)d_in[15];
    const float* bs1  = (const float*)d_in[16];
    const float* Ws2  = (const float*)d_in[17];
    const float* bs2  = (const float*)d_in[18];

    const int N = in_sizes[0] / H;     // 10000 (multiple of 16)
    const int E = in_sizes[2] / 2;     // 320000 (multiple of ET)
    const int* tgt = ei;
    const int* src = ei + E;

    // ---- workspace layout ----
    _Float16* Wi1p = (_Float16*)d_ws;
    _Float16* Wi2p = Wi1p + 128 * 384;
    _Float16* Wf2p = Wi2p + 384 * 384;
    _Float16* Wvp  = Wf2p + 128 * 384;
    _Float16* Ws1p = Wvp  + 128 * 256;
    _Float16* Ws2p = Ws1p + 256 * 384;
    float* fbase  = (float*)(Ws2p + 384 * 384);
    const size_t szH = (size_t)N * H3;
    float* bufH   = fbase;                           // h1 / hs1
    float* bufX   = bufH + szH;                      // x  / delta
    float* qmsg   = bufX + szH;                      // [N,H]   (later: inner)
    float* mumsg  = qmsg + (size_t)N * H;            // [N,3,H]
    float* deg    = mumsg + (size_t)N * H3;          // [N] (padded)
    const size_t degPad = (size_t)((N + 63) & ~63);
    float* mu_cat = deg + degPad;                    // [3N,256]
    float* sinb   = mu_cat + (size_t)N * 3 * 256;    // [N,256]
    float* inner  = qmsg;                            // reuse after edge phase
    float* qn  = (float*)d_out;                      // [N,128]
    float* mun = qn + (size_t)N * H;                 // [N,3,128]

    // ---- zero the atomic accumulators (contiguous: qmsg|mumsg|deg) ----
    const size_t zcount = (size_t)N * H + (size_t)N * H3 + degPad;
    zero_kernel<<<(zcount + 255) / 256, 256, 0, stream>>>(qmsg, zcount);

    // ---- pack weights to WMMA fragment order (f16) ----
    pack_b_kernel<<<(384 / 16) * (128 / 32), 32, 0, stream>>>(Wi1, Wi1p, 128, 384);
    pack_b_kernel<<<(384 / 16) * (384 / 32), 32, 0, stream>>>(Wi2, Wi2p, 384, 384);
    pack_b_kernel<<<(384 / 16) * (128 / 32), 32, 0, stream>>>(Wf2, Wf2p, 128, 384);
    pack_b_kernel<<<(256 / 16) * (128 / 32), 32, 0, stream>>>(Wv,  Wvp,  128, 256);
    pack_b_kernel<<<(384 / 16) * (256 / 32), 32, 0, stream>>>(Ws1, Ws1p, 256, 384);
    pack_b_kernel<<<(384 / 16) * (384 / 32), 32, 0, stream>>>(Ws2, Ws2p, 384, 384);

    // ---- node interaction MLP: x = silu(q@Wi1+bi1)@Wi2+bi2 ----
    dim3 gN(N / 16, 384 / 64);
    wmma_gemm_kernel<1><<<gN, 128, 0, stream>>>(q,    Wi1p, bi1, bufH, N, 384, 128);
    wmma_gemm_kernel<0><<<gN, 128, 0, stream>>>(bufH, Wi2p, bi2, bufX, N, 384, 384);

    // ---- fused filter-network + edge messages ----
    edge_kernel<<<E / ET, 128, 0, stream>>>(bufX, mu, tgt, src, rbf, unit, cut,
                                            Wf1, bf1, Wf2p, bf2, qmsg, mumsg, deg);

    // ---- q_new / mu_new -> d_out ----
    const size_t fin = (size_t)N * (H + H3);
    finalize_nodes_kernel<<<(fin + 255) / 256, 256, 0, stream>>>(q, mu, qmsg, mumsg,
                                                                 deg, qn, mun, N);

    // ---- mixing: mu_cat = mu_new @ Wv (rows = (n,c)) ----
    dim3 gV(3 * N / 16, 256 / 64);
    wmma_gemm_kernel<0><<<gV, 128, 0, stream>>>(mun, Wvp, nullptr, mu_cat, 3 * N, 256, 128);

    mix_reduce_kernel<<<(N * H + 255) / 256, 256, 0, stream>>>(qn, mu_cat, sinb, inner, N);

    // ---- scalar MLP: delta = silu(sin@Ws1+bs1)@Ws2+bs2 ----
    wmma_gemm_kernel<1><<<gN, 128, 0, stream>>>(sinb, Ws1p, bs1, bufH, N, 384, 256);
    wmma_gemm_kernel<0><<<gN, 128, 0, stream>>>(bufH, Ws2p, bs2, bufX, N, 384, 384);

    final_update_kernel<<<(N * H + 255) / 256, 256, 0, stream>>>(qn, mun, bufX,
                                                                 mu_cat, inner, N);
}
